// BiWindowMambaLayer_46428596469827
// MI455X (gfx1250) — matrix operand order repaired and checked by reading, not compiled, not run on hardware
//
#include <hip/hip_runtime.h>
#include <hip/hip_bf16.h>
#include <hip/hip_fp16.h>

typedef _Float16 v16h __attribute__((ext_vector_type(16)));
typedef _Float16 v8h  __attribute__((ext_vector_type(8)));
typedef float    v8f  __attribute__((ext_vector_type(8)));

#define DIMC   256
#define L_SEQ  1024
#define BATCH  2
#define M_TOT  2048      // BATCH * L_SEQ
#define D_IN   512
#define D_ST   16
#define DT_R   16
#define NCH    16        // scan chunks per sequence
#define CHL    64        // chunk length (NCH*CHL == L_SEQ)

__device__ __forceinline__ float siluf(float v) { return v / (1.0f + __expf(-v)); }
__device__ __forceinline__ float softplusf(float v) {
    return (v > 20.0f) ? v : log1pf(__expf(v));
}

// ---------------- f32 -> f16 convert ----------------
__global__ void k_f32_to_f16(const float* __restrict__ src, _Float16* __restrict__ dst, int n) {
    int i = blockIdx.x * blockDim.x + threadIdx.x;
    if (i < n) dst[i] = (_Float16)src[i];
}

// ---------------- avgpool 2x2 + layernorm -> f16 ----------------
__global__ void k_pool_ln(const float* __restrict__ x, const float* __restrict__ g,
                          const float* __restrict__ beta, _Float16* __restrict__ xn) {
    int m = blockIdx.x;
    int c = threadIdx.x;
    int b = m >> 10, l = m & 1023;
    int nh = l >> 5, nw = l & 31;
    const float* px = x + (((size_t)(b * DIMC + c)) << 12);   // * 64*64
    int h = nh * 2, w = nw * 2;
    float v = 0.25f * (px[h * 64 + w] + px[h * 64 + w + 1] +
                       px[(h + 1) * 64 + w] + px[(h + 1) * 64 + w + 1]);
    __shared__ float s1[256], s2[256];
    s1[c] = v; s2[c] = v * v;
    __syncthreads();
    for (int s = 128; s > 0; s >>= 1) {
        if (c < s) { s1[c] += s1[c + s]; s2[c] += s2[c + s]; }
        __syncthreads();
    }
    float mean = s1[0] * (1.0f / 256.0f);
    float var  = s2[0] * (1.0f / 256.0f) - mean * mean;
    float r    = rsqrtf(var + 1e-5f);
    xn[(size_t)m * DIMC + c] = (_Float16)((v - mean) * r * g[c] + beta[c]);
}

// ---------------- WMMA GEMM: C[M,N] = A[M,K] * B[N,K]^T, f16 in, f32 out ----
// one wave per 16x64 strip: A fragment reused across 4 B tiles (4 accumulators)
__global__ void k_gemm_wmma(const _Float16* __restrict__ A, const _Float16* __restrict__ B,
                            float* __restrict__ C, int M, int N, int K) {
    int lane = threadIdx.x & 31;
    int nt = blockIdx.x;              // 64-wide strip index
    int mt = blockIdx.y;              // 16-row tile index
    int hs  = lane >> 4;              // half-wave select
    int r16 = lane & 15;
    const _Float16* arow = A + ((size_t)(mt * 16 + r16)) * K;
    const _Float16* brow[4];
#pragma unroll
    for (int j = 0; j < 4; ++j)
        brow[j] = B + ((size_t)(nt * 64 + j * 16 + r16)) * K;
    v8f acc0 = {}, acc1 = {}, acc2 = {}, acc3 = {};
    for (int kb = 0; kb < K; kb += 32) {
        // A 16x32: lanes 0-15 row r16 hold K kb..kb+15; lanes 16-31 kb+16..kb+31
        int ka = kb + hs * 16;
        v8h alo = *(const v8h*)(arow + ka);
        v8h ahi = *(const v8h*)(arow + ka + 8);
        v16h av = __builtin_shufflevector(alo, ahi, 0, 1, 2, 3, 4, 5, 6, 7,
                                                    8, 9, 10, 11, 12, 13, 14, 15);
        // B 32x16: lanes 0-15 col r16 hold K kb..kb+15; lanes 16-31 kb+16..kb+31
        int kboff = kb + hs * 16;
        v16h bv0 = *(const v16h*)(brow[0] + kboff);
        v16h bv1 = *(const v16h*)(brow[1] + kboff);
        v16h bv2 = *(const v16h*)(brow[2] + kboff);
        v16h bv3 = *(const v16h*)(brow[3] + kboff);
        acc0 = __builtin_amdgcn_wmma_f32_16x16x32_f16(false, av, false, bv0, (short)0, acc0, false, false);
        acc1 = __builtin_amdgcn_wmma_f32_16x16x32_f16(false, av, false, bv1, (short)0, acc1, false, false);
        acc2 = __builtin_amdgcn_wmma_f32_16x16x32_f16(false, av, false, bv2, (short)0, acc2, false, false);
        acc3 = __builtin_amdgcn_wmma_f32_16x16x32_f16(false, av, false, bv3, (short)0, acc3, false, false);
    }
    int orow = mt * 16 + hs * 8;
    float* co = C + (size_t)orow * N + nt * 64 + r16;
#pragma unroll
    for (int v = 0; v < 8; ++v) {
        co[(size_t)v * N +  0] = acc0[v];
        co[(size_t)v * N + 16] = acc1[v];
        co[(size_t)v * N + 32] = acc2[v];
        co[(size_t)v * N + 48] = acc3[v];
    }
}

// ---------------- depthwise causal/anticausal conv + bias + SiLU ----------------
__global__ void k_conv_silu(const float* __restrict__ xz,
                            const float* __restrict__ fw, const float* __restrict__ fb,
                            const float* __restrict__ bw, const float* __restrict__ bb,
                            float* __restrict__ u) {
    int gid = blockIdx.x * blockDim.x + threadIdx.x;
    if (gid >= 2 * M_TOT * D_IN) return;
    int dir = gid / (M_TOT * D_IN);
    int rem = gid - dir * (M_TOT * D_IN);
    int m = rem / D_IN;
    int d = rem - m * D_IN;
    int b = m >> 10, l = m & 1023;
    const float* w = (dir ? bw : fw) + d * 4;
    float acc = (dir ? bb : fb)[d];
    size_t base = ((size_t)(b << 10)) * 2048 + (size_t)(dir * 1024 + d);
#pragma unroll
    for (int j = 0; j < 4; ++j) {
        int t = dir ? (l + 3 - j) : (l - 3 + j);
        if (t >= 0 && t < L_SEQ) acc += w[j] * xz[base + (size_t)t * 2048];
    }
    u[gid] = siluf(acc);
}

// ---------------- x_proj (48 outs) + dt_proj + softplus ----------------
__global__ void k_xproj_dt(const float* __restrict__ u,
                           const float* __restrict__ f_xw, const float* __restrict__ b_xw,
                           const float* __restrict__ f_dw, const float* __restrict__ f_db,
                           const float* __restrict__ b_dw, const float* __restrict__ b_db,
                           float* __restrict__ Bm, float* __restrict__ Cm,
                           float* __restrict__ delta) {
    int blk = blockIdx.x;                 // dir*2048 + m
    int dir = blk >> 11;
    int m   = blk & 2047;
    int t   = threadIdx.x;                // 0..255
    __shared__ float su[D_IN];
    __shared__ float sdbl[48];
    size_t idx = (size_t)dir * M_TOT + m;
    const float* urow = u + idx * D_IN;
    su[t] = urow[t];
    su[t + 256] = urow[t + 256];
    __syncthreads();
    const float* xw = dir ? b_xw : f_xw;
    if (t < 48) {
        const float* wr = xw + t * D_IN;
        float s = 0.0f;
        for (int k = 0; k < D_IN; ++k) s += su[k] * wr[k];
        sdbl[t] = s;
        if (t >= 16 && t < 32) Bm[idx * D_ST + (t - 16)] = s;
        if (t >= 32)           Cm[idx * D_ST + (t - 32)] = s;
    }
    __syncthreads();
    const float* dw = dir ? b_dw : f_dw;
    const float* db = dir ? b_db : f_db;
    for (int d = t; d < D_IN; d += 256) {
        float a = db[d];
        const float* wr = dw + d * DT_R;
#pragma unroll
        for (int r = 0; r < DT_R; ++r) a += sdbl[r] * wr[r];
        delta[idx * D_IN + d] = softplusf(a);
    }
}

// =========================== chunked selective scan =========================
// recurrence h <- a*h + b is elementwise per (d,n); over a chunk:
// h_out = P*h_in + S with P = prod(a), S = sum(prod(later a)*b).
// time index t = c*CHL + i; position p = dir ? L-1-t : t.

// Phase 1: per-chunk transition (P, S) with h_in = 0.  grid = 2*2*NCH blocks.
__global__ void k_scan_p1(const float* __restrict__ u, const float* __restrict__ delta,
                          const float* __restrict__ Bm,
                          const float* __restrict__ f_Alog, const float* __restrict__ b_Alog,
                          float* __restrict__ chunkP, float* __restrict__ chunkS) {
    int blk = blockIdx.x;                       // dir*32 + b*16 + c
    int dir = blk >> 5;
    int b   = (blk >> 4) & 1;
    int c   = blk & 15;
    int d   = threadIdx.x;                      // 0..511
    const float* Alog = (dir ? b_Alog : f_Alog) + d * D_ST;
    float A[D_ST], P[D_ST], S[D_ST];
#pragma unroll
    for (int n = 0; n < D_ST; ++n) { A[n] = -__expf(Alog[n]); P[n] = 1.0f; S[n] = 0.0f; }
    for (int i = 0; i < CHL; ++i) {
        int t = c * CHL + i;
        int p = dir ? (L_SEQ - 1 - t) : t;
        int m = (b << 10) + p;
        size_t idx = (size_t)dir * M_TOT + m;
        float dv = delta[idx * D_IN + d];
        float uv = u[idx * D_IN + d];
        float du = dv * uv;
        const float* Bp = Bm + idx * D_ST;
#pragma unroll
        for (int n = 0; n < D_ST; ++n) {
            float a = __expf(dv * A[n]);
            S[n] = a * S[n] + du * Bp[n];
            P[n] *= a;
        }
    }
    size_t base = ((size_t)blk * D_IN + d) * D_ST;
#pragma unroll
    for (int n = 0; n < D_ST; ++n) { chunkP[base + n] = P[n]; chunkS[base + n] = S[n]; }
}

// Phase 2: compose chunk transitions sequentially (16 steps). grid = 4 blocks.
__global__ void k_scan_p2(const float* __restrict__ chunkP, const float* __restrict__ chunkS,
                          float* __restrict__ chunkH) {
    int dir = blockIdx.x >> 1, b = blockIdx.x & 1;
    int d = threadIdx.x;
    float h[D_ST];
#pragma unroll
    for (int n = 0; n < D_ST; ++n) h[n] = 0.0f;
    for (int c = 0; c < NCH; ++c) {
        size_t base = ((size_t)(dir * 32 + b * 16 + c) * D_IN + d) * D_ST;
#pragma unroll
        for (int n = 0; n < D_ST; ++n) {
            chunkH[base + n] = h[n];
            h[n] = chunkP[base + n] * h[n] + chunkS[base + n];
        }
    }
}

// Phase 3: replay each chunk from correct h_in, emit gated outputs. grid = 64.
__global__ void k_scan_p3(const float* __restrict__ u, const float* __restrict__ delta,
                          const float* __restrict__ Bm, const float* __restrict__ Cm,
                          const float* __restrict__ xz, const float* __restrict__ chunkH,
                          const float* __restrict__ f_Alog, const float* __restrict__ f_D,
                          const float* __restrict__ b_Alog, const float* __restrict__ b_D,
                          float* __restrict__ yf, float* __restrict__ yb) {
    int blk = blockIdx.x;
    int dir = blk >> 5;
    int b   = (blk >> 4) & 1;
    int c   = blk & 15;
    int d   = threadIdx.x;
    const float* Alog = (dir ? b_Alog : f_Alog) + d * D_ST;
    float Dv = (dir ? b_D : f_D)[d];
    float A[D_ST], h[D_ST];
    size_t base = ((size_t)blk * D_IN + d) * D_ST;
#pragma unroll
    for (int n = 0; n < D_ST; ++n) { A[n] = -__expf(Alog[n]); h[n] = chunkH[base + n]; }
    float* yout = dir ? yb : yf;
    for (int i = 0; i < CHL; ++i) {
        int t = c * CHL + i;
        int p = dir ? (L_SEQ - 1 - t) : t;
        int m = (b << 10) + p;
        size_t idx = (size_t)dir * M_TOT + m;
        float dv = delta[idx * D_IN + d];
        float uv = u[idx * D_IN + d];
        float du = dv * uv;
        const float* Bp = Bm + idx * D_ST;
        const float* Cp = Cm + idx * D_ST;
        float y = 0.0f;
#pragma unroll
        for (int n = 0; n < D_ST; ++n) {
            float hn = __expf(dv * A[n]) * h[n] + du * Bp[n];
            h[n] = hn;
            y += hn * Cp[n];
        }
        float zv = xz[(size_t)m * 2048 + dir * 1024 + 512 + d];
        yout[(size_t)m * D_IN + d] = (y + uv * Dv) * siluf(zv);
    }
}

// ---------------- yf + yb -> f16 ----------------
__global__ void k_pack(const float* __restrict__ yf, const float* __restrict__ yb,
                       _Float16* __restrict__ yh, int n) {
    int i = blockIdx.x * blockDim.x + threadIdx.x;
    if (i < n) yh[i] = (_Float16)(yf[i] + yb[i]);
}

// ---------------- nearest upsample x2 + residual ----------------
__global__ void k_upsample_res(const float* __restrict__ xo, const float* __restrict__ x,
                               float* __restrict__ out) {
    int i = blockIdx.x * blockDim.x + threadIdx.x;
    if (i >= BATCH * DIMC * 64 * 64) return;
    int w = i & 63, h = (i >> 6) & 63, c = (i >> 12) & 255, b = i >> 20;
    int l = (h >> 1) * 32 + (w >> 1);
    int m = (b << 10) + l;
    out[i] = xo[(size_t)m * DIMC + c] + x[i];
}

extern "C" void kernel_launch(void* const* d_in, const int* in_sizes, int n_in,
                              void* d_out, int out_size, void* d_ws, size_t ws_size,
                              hipStream_t stream) {
    const float* x        = (const float*)d_in[0];
    const float* ln_g     = (const float*)d_in[1];
    const float* ln_b     = (const float*)d_in[2];
    const float* f_in_w   = (const float*)d_in[3];
    const float* f_conv_w = (const float*)d_in[4];
    const float* f_conv_b = (const float*)d_in[5];
    const float* f_xproj  = (const float*)d_in[6];
    const float* f_dt_w   = (const float*)d_in[7];
    const float* f_dt_b   = (const float*)d_in[8];
    const float* f_A_log  = (const float*)d_in[9];
    const float* f_Dp     = (const float*)d_in[10];
    const float* b_in_w   = (const float*)d_in[11];
    const float* b_conv_w = (const float*)d_in[12];
    const float* b_conv_b = (const float*)d_in[13];
    const float* b_xproj  = (const float*)d_in[14];
    const float* b_dt_w   = (const float*)d_in[15];
    const float* b_dt_b   = (const float*)d_in[16];
    const float* b_A_log  = (const float*)d_in[17];
    const float* b_Dp     = (const float*)d_in[18];
    const float* out_w    = (const float*)d_in[19];
    float* out = (float*)d_out;

    // workspace carving
    char* p = (char*)d_ws;
    _Float16* xn_h   = (_Float16*)p;  p += (size_t)M_TOT * DIMC * 2;        // 1 MB
    _Float16* w_in_h = (_Float16*)p;  p += (size_t)2048 * DIMC * 2;         // 1 MB
    _Float16* w_out_h= (_Float16*)p;  p += (size_t)DIMC * D_IN * 2;         // 256 KB
    float* xz    = (float*)p;         p += (size_t)M_TOT * 2048 * 4;        // 16 MB
    float* u     = (float*)p;         p += (size_t)2 * M_TOT * D_IN * 4;    // 8 MB
    float* delta = (float*)p;         p += (size_t)2 * M_TOT * D_IN * 4;    // 8 MB
    float* Bm    = (float*)p;         p += (size_t)2 * M_TOT * D_ST * 4;    // 256 KB
    float* Cm    = (float*)p;         p += (size_t)2 * M_TOT * D_ST * 4;    // 256 KB
    float* yf    = (float*)p;         p += (size_t)M_TOT * D_IN * 4;        // 4 MB
    float* yb    = (float*)p;         p += (size_t)M_TOT * D_IN * 4;        // 4 MB
    _Float16* y_h = (_Float16*)p;     p += (size_t)M_TOT * D_IN * 2;        // 2 MB
    float* xo    = (float*)p;         p += (size_t)M_TOT * DIMC * 4;        // 2 MB
    size_t chsz = (size_t)2 * BATCH * NCH * D_IN * D_ST;                    // 512K floats
    float* chunkP = (float*)p;        p += chsz * 4;                        // 2 MB
    float* chunkS = (float*)p;        p += chsz * 4;                        // 2 MB
    float* chunkH = (float*)p;        p += chsz * 4;                        // 2 MB

    // 1) weight conversion to f16 (fwd rows 0..1023, bwd rows 1024..2047)
    {
        int n1 = 1024 * DIMC;
        k_f32_to_f16<<<(n1 + 255) / 256, 256, 0, stream>>>(f_in_w, w_in_h, n1);
        k_f32_to_f16<<<(n1 + 255) / 256, 256, 0, stream>>>(b_in_w, w_in_h + n1, n1);
        int n2 = DIMC * D_IN;
        k_f32_to_f16<<<(n2 + 255) / 256, 256, 0, stream>>>(out_w, w_out_h, n2);
    }
    // 2) avgpool + layernorm -> f16 tokens
    k_pool_ln<<<M_TOT, 256, 0, stream>>>(x, ln_g, ln_b, xn_h);
    // 3) fused fwd+bwd in_proj GEMM: (2048 x 256) x (256 x 2048) via WMMA, 16x64 strips
    k_gemm_wmma<<<dim3(2048 / 64, M_TOT / 16), 32, 0, stream>>>(
        xn_h, w_in_h, xz, M_TOT, 2048, DIMC);
    // 4) depthwise conv + SiLU
    k_conv_silu<<<(2 * M_TOT * D_IN) / 256, 256, 0, stream>>>(
        xz, f_conv_w, f_conv_b, b_conv_w, b_conv_b, u);
    // 5) x_proj + dt_proj + softplus
    k_xproj_dt<<<2 * M_TOT, 256, 0, stream>>>(
        u, f_xproj, b_xproj, f_dt_w, f_dt_b, b_dt_w, b_dt_b, Bm, Cm, delta);
    // 6) chunked selective scan: 16x more parallelism than naive sequential
    k_scan_p1<<<2 * BATCH * NCH, D_IN, 0, stream>>>(
        u, delta, Bm, f_A_log, b_A_log, chunkP, chunkS);
    k_scan_p2<<<2 * BATCH, D_IN, 0, stream>>>(chunkP, chunkS, chunkH);
    k_scan_p3<<<2 * BATCH * NCH, D_IN, 0, stream>>>(
        u, delta, Bm, Cm, xz, chunkH, f_A_log, f_Dp, b_A_log, b_Dp, yf, yb);
    // 7) yf + yb -> f16
    k_pack<<<(M_TOT * D_IN) / 256, 256, 0, stream>>>(yf, yb, y_h, M_TOT * D_IN);
    // 8) out_proj GEMM: (2048 x 512) x (512 x 256) via WMMA
    k_gemm_wmma<<<dim3(DIMC / 64, M_TOT / 16), 32, 0, stream>>>(
        y_h, w_out_h, xo, M_TOT, DIMC, D_IN);
    // 9) upsample + residual
    k_upsample_res<<<(BATCH * DIMC * 64 * 64) / 256, 256, 0, stream>>>(xo, x, out);
}
